// MoELayer_4681514353281
// MI455X (gfx1250) — matrix operand order repaired and checked by reading, not compiled, not run on hardware
//
#include <hip/hip_runtime.h>
#include <hip/hip_bf16.h>
#include <math.h>

// ---------- types ----------
typedef __attribute__((ext_vector_type(16))) __bf16          v16bf;
typedef __attribute__((ext_vector_type(8)))  float           v8f;
typedef __attribute__((ext_vector_type(8)))  unsigned short  su8;   // 128-bit chunk of 8 bf16

union FragU { su8 h[2]; v16bf bf; };

__device__ __forceinline__ unsigned short f32_to_bf16_rne(float f) {
  unsigned int u = __float_as_uint(f);
  u += 0x7FFFu + ((u >> 16) & 1u);      // round-to-nearest-even
  return (unsigned short)(u >> 16);
}

__device__ __forceinline__ float gelu_tanh(float x) {
  const float k0 = 0.7978845608028654f; // sqrt(2/pi)
  const float k1 = 0.044715f;
  float t = tanhf(k0 * (x + k1 * x * x * x));
  return 0.5f * x * (1.0f + t);
}

// ---------- CDNA5 async global->LDS copy (ASYNCcnt-tracked, bypasses VGPRs) ----------
// LDS byte address = low 32 bits of the generic shared-memory address (ISA: LDS aperture
// maps addr[31:0] -> LDS offset).
__device__ __forceinline__ void async_copy_b128(void* lds_ptr, const void* gptr) {
  unsigned loff = (unsigned)(size_t)lds_ptr;
  asm volatile("global_load_async_to_lds_b128 %0, %1, off"
               :: "v"(loff), "v"(gptr)
               : "memory");
}
__device__ __forceinline__ void wait_async0() {
  asm volatile("s_wait_asynccnt 0x0" ::: "memory");
}

// ---------- simple utility kernels ----------
__global__ __launch_bounds__(256) void zero_kernel(float* __restrict__ p, size_t n) {
  size_t i = (size_t)blockIdx.x * blockDim.x + threadIdx.x;
  if (i < n) p[i] = 0.0f;
}

__global__ __launch_bounds__(256) void cast_bf16_kernel(const float* __restrict__ x,
                                                        unsigned short* __restrict__ y,
                                                        size_t n) {
  size_t i = (size_t)blockIdx.x * blockDim.x + threadIdx.x;
  if (i < n) y[i] = f32_to_bf16_rne(x[i]);
}

// ---------- gating: softmax over 16 logits, top-4, renormalize ----------
// wgt[n*4 + e] = routing weight of expert e (e in 0..3) for token n, 0 if not selected.
__global__ __launch_bounds__(256) void gate_kernel(const float* __restrict__ x,
                                                   const float* __restrict__ Wg,
                                                   float* __restrict__ wgt,
                                                   int N, int D) {
  int n = blockIdx.x * blockDim.x + threadIdx.x;
  if (n >= N) return;
  const float* xr = x + (size_t)n * D;
  float logit[16];
#pragma unroll
  for (int e = 0; e < 16; ++e) logit[e] = 0.0f;
  for (int d = 0; d < D; ++d) {
    float xv = xr[d];
    const float* wr = Wg + (size_t)d * 16;
#pragma unroll
    for (int e = 0; e < 16; ++e) logit[e] += xv * wr[e];
  }
  float m = logit[0];
#pragma unroll
  for (int e = 1; e < 16; ++e) m = fmaxf(m, logit[e]);
  float p[16]; float sum = 0.0f;
#pragma unroll
  for (int e = 0; e < 16; ++e) { p[e] = expf(logit[e] - m); sum += p[e]; }
  float inv = 1.0f / sum;
#pragma unroll
  for (int e = 0; e < 16; ++e) p[e] *= inv;

  float tw[4]; int ti[4];
#pragma unroll
  for (int k = 0; k < 4; ++k) {
    float best = -1.0f; int bi = 0;
#pragma unroll
    for (int e = 0; e < 16; ++e) { if (p[e] > best) { best = p[e]; bi = e; } }
    tw[k] = best; ti[k] = bi; p[bi] = -1.0f;
  }
  float s4 = tw[0] + tw[1] + tw[2] + tw[3];
  float w4[4] = {0.f, 0.f, 0.f, 0.f};
#pragma unroll
  for (int k = 0; k < 4; ++k) if (ti[k] < 4) w4[ti[k]] = tw[k] / s4;
#pragma unroll
  for (int i = 0; i < 4; ++i) wgt[(size_t)n * 4 + i] = w4[i];
}

// ---------- fold LoRA into dense weight, transpose, cast bf16 ----------
// Input:  W [K][M] f32, A [K][16] f32, B [16][M] f32  (per expert, strided by blockIdx.z)
// Output: Wt [M][K] bf16 = transpose(W + A@B)   -> contiguous K for WMMA B-fragments
__global__ __launch_bounds__(256) void prep_weight_kernel(
    const float* __restrict__ W, const float* __restrict__ A, const float* __restrict__ B,
    unsigned short* __restrict__ Wt, int K, int M,
    size_t wStride, size_t aStride, size_t bStride, size_t oStride) {
  int e = blockIdx.z;
  W += e * wStride; A += e * aStride; B += e * bStride; Wt += e * oStride;

  __shared__ float tile[32][33];
  const int k0 = blockIdx.x * 32;
  const int m0 = blockIdx.y * 32;
  const int tm = threadIdx.x & 31;   // m within tile (coalesced reads)
  const int tq = threadIdx.x >> 5;   // 0..7

  for (int kk = tq; kk < 32; kk += 8) {
    const int k = k0 + kk, m = m0 + tm;
    float v = W[(size_t)k * M + m];
#pragma unroll
    for (int r = 0; r < 16; ++r) v += A[(size_t)k * 16 + r] * B[(size_t)r * M + m];
    tile[kk][tm] = v;
  }
  __syncthreads();
  const int tk = threadIdx.x & 31;   // k within tile (coalesced writes)
  for (int mm = tq; mm < 32; mm += 8) {
    Wt[(size_t)(m0 + mm) * K + k0 + tk] = f32_to_bf16_rne(tile[tk][mm]);
  }
}

// ---------- WMMA GEMM: C[M,N] = A[M,K] * Bt[N,K]^T ----------
// EPI==0 : Hout = bf16(gelu(C))          (GEMM1)
// EPI==1 : Out += wgt[row,expert] * C    (GEMM2, out accumulate)
#define BM 128
#define BN 128
#define BK 32

template <int EPI>
__global__ __launch_bounds__(256, 1)
void moe_gemm_kernel(const unsigned short* __restrict__ Abuf,   // [Mrows][Kdim] bf16
                     const unsigned short* __restrict__ Bbuf,   // [Ncols][Kdim] bf16
                     unsigned short* __restrict__ Hout,         // EPI==0
                     float* __restrict__ Out,                   // EPI==1
                     const float* __restrict__ wgt,             // [Mrows][4]
                     int expert, int Kdim, int Ncols) {
  __shared__ unsigned short sA[2][BM * BK];
  __shared__ unsigned short sB[2][BN * BK];

  const int tid  = threadIdx.x;
  const int lane = tid & 31;
  const int wave = tid >> 5;
  const int wm   = wave >> 2;            // 0..1 -> 64-row strip
  const int wn   = wave & 3;             // 0..3 -> 32-col strip
  const int m0   = blockIdx.y * BM;
  const int n0   = blockIdx.x * BN;

  // cooperative tile copy mapping: each thread moves 32B of A and 32B of B per tile
  const int ldRow = tid >> 1;            // 0..127
  const int ldCol = (tid & 1) << 4;      // 0 or 16
  const unsigned short* gA = Abuf + (size_t)(m0 + ldRow) * Kdim + ldCol;
  const unsigned short* gB = Bbuf + (size_t)(n0 + ldRow) * Kdim + ldCol;
  const int sOff = ldRow * BK + ldCol;

  v8f acc[4][2];
#pragma unroll
  for (int i = 0; i < 4; ++i)
#pragma unroll
    for (int j = 0; j < 2; ++j)
#pragma unroll
      for (int v = 0; v < 8; ++v) acc[i][j][v] = 0.0f;

  // issue one K-step tile as 4 async 128-bit L2->LDS copies (ASYNCcnt)
  auto issue_tile = [&](int ks, int buf) {
    const unsigned short* pa = gA + (size_t)ks * BK;
    const unsigned short* pb = gB + (size_t)ks * BK;
    async_copy_b128(&sA[buf][sOff],     pa);
    async_copy_b128(&sA[buf][sOff + 8], pa + 8);
    async_copy_b128(&sB[buf][sOff],     pb);
    async_copy_b128(&sB[buf][sOff + 8], pb + 8);
  };

  issue_tile(0, 0);
  wait_async0();
  __syncthreads();

  // per-lane fragment addressing (ISA 7.12.2 layouts)
  const int aK = (lane < 16) ? 0 : 8;    // A: K{0..7,16..23} | K{8..15,24..31}
  const int bK = (lane < 16) ? 0 : 8;    // B: K{0..7,16..23} | K{8..15,24..31}
  const int aRowBase = wm * 64 + (lane & 15);
  const int bColBase = wn * 32 + (lane & 15);

  const int steps = Kdim / BK;
  for (int ks = 0; ks < steps; ++ks) {
    const int buf = ks & 1;
    // prefetch next K-step into the other buffer; safe: all waves passed the
    // previous barrier, so all reads of buf^1 (iteration ks-1) are complete.
    if (ks + 1 < steps) issue_tile(ks + 1, buf ^ 1);

    FragU fa[4], fb[2];
#pragma unroll
    for (int i = 0; i < 4; ++i) {
      const int r = (aRowBase + i * 16) * BK;
      fa[i].h[0] = *(const su8*)&sA[buf][r + aK];
      fa[i].h[1] = *(const su8*)&sA[buf][r + aK + 16];
    }
#pragma unroll
    for (int j = 0; j < 2; ++j) {
      const int c = (bColBase + j * 16) * BK;
      fb[j].h[0] = *(const su8*)&sB[buf][c + bK];
      fb[j].h[1] = *(const su8*)&sB[buf][c + bK + 16];
    }
#pragma unroll
    for (int i = 0; i < 4; ++i)
#pragma unroll
      for (int j = 0; j < 2; ++j)
        acc[i][j] = __builtin_amdgcn_wmma_f32_16x16x32_bf16(
            false, fa[i].bf, false, fb[j].bf, (short)0, acc[i][j], false, false);

    wait_async0();       // prefetched buffer fully written to LDS
    __syncthreads();     // visible to all waves before next iteration reads it
  }

  // epilogue: C VGPR v holds M = v + (lane<16?0:8), N = lane%16
  const int rowSel = (lane < 16) ? 0 : 8;
  const int colIn  = lane & 15;
#pragma unroll
  for (int i = 0; i < 4; ++i) {
    const int rbase = m0 + wm * 64 + i * 16 + rowSel;
#pragma unroll
    for (int j = 0; j < 2; ++j) {
      const int col = n0 + wn * 32 + j * 16 + colIn;
#pragma unroll
      for (int v = 0; v < 8; ++v) {
        const size_t o = (size_t)(rbase + v) * Ncols + col;
        if (EPI == 0) {
          Hout[o] = f32_to_bf16_rne(gelu_tanh(acc[i][j][v]));
        } else {
          const float wv = wgt[(size_t)(rbase + v) * 4 + expert];
          Out[o] += wv * acc[i][j][v];
        }
      }
    }
  }
}

// ---------- launcher ----------
extern "C" void kernel_launch(void* const* d_in, const int* in_sizes, int n_in,
                              void* d_out, int out_size, void* d_ws, size_t ws_size,
                              hipStream_t stream) {
  (void)in_sizes; (void)n_in; (void)out_size; (void)ws_size;
  const float* x  = (const float*)d_in[0];
  const float* Wg = (const float*)d_in[1];
  const float* W1 = (const float*)d_in[2];
  const float* A1 = (const float*)d_in[3];
  const float* B1 = (const float*)d_in[4];
  const float* W2 = (const float*)d_in[5];
  const float* A2 = (const float*)d_in[6];
  const float* B2 = (const float*)d_in[7];
  float* out = (float*)d_out;

  const int N = 2 * 4096;   // tokens
  const int D = 1024;
  const int F = 4096;

  // workspace layout (bytes)
  uint8_t* ws = (uint8_t*)d_ws;
  unsigned short* xb  = (unsigned short*)(ws);                       // N*D bf16   (16.0 MiB)
  float*          wgt = (float*)(ws + (size_t)16777216);             // N*4 f32    (0.125 MiB)
  unsigned short* Wt1 = (unsigned short*)(ws + (size_t)16908288);    // 4*F*D bf16 (32 MiB)
  unsigned short* Wt2 = (unsigned short*)(ws + (size_t)50462720);    // 4*D*F bf16 (32 MiB)
  unsigned short* H   = (unsigned short*)(ws + (size_t)84017152);    // N*F bf16   (64 MiB)

  const size_t nOut = (size_t)N * D;
  zero_kernel<<<(nOut + 255) / 256, 256, 0, stream>>>(out, nOut);
  cast_bf16_kernel<<<(nOut + 255) / 256, 256, 0, stream>>>(x, xb, nOut);
  gate_kernel<<<(N + 255) / 256, 256, 0, stream>>>(x, Wg, wgt, N, D);

  // Wt1[e][f][d] = (W1[e] + A1[e]@B1[e])^T : K=D rows, M=F cols
  prep_weight_kernel<<<dim3(D / 32, F / 32, 4), 256, 0, stream>>>(
      W1, A1, B1, Wt1, D, F,
      (size_t)D * F, (size_t)D * 16, (size_t)16 * F, (size_t)F * D);
  // Wt2[e][d][f] = (W2[e] + A2[e]@B2[e])^T : K=F rows, M=D cols
  prep_weight_kernel<<<dim3(F / 32, D / 32, 4), 256, 0, stream>>>(
      W2, A2, B2, Wt2, F, D,
      (size_t)F * D, (size_t)F * 16, (size_t)16 * D, (size_t)D * F);

  for (int e = 0; e < 4; ++e) {
    moe_gemm_kernel<0><<<dim3(F / BN, N / BM), 256, 0, stream>>>(
        xb, Wt1 + (size_t)e * F * D, H, nullptr, nullptr, e, D, F);
    moe_gemm_kernel<1><<<dim3(D / BN, N / BM), 256, 0, stream>>>(
        H, Wt2 + (size_t)e * D * F, nullptr, out, wgt, e, F, D);
  }
}
